// GCN_Contrastive_49692771615405
// MI455X (gfx1250) — compile-verified, matrix operand order checked by reading, not run on hardware
//
#include <hip/hip_runtime.h>
#include <hip/hip_bf16.h>

typedef __attribute__((ext_vector_type(16))) _Float16 v16h;
typedef __attribute__((ext_vector_type(4)))  _Float16 v4h;
typedef __attribute__((ext_vector_type(8)))  float    v8f;

#define N_NODES 100000

// ---------------------------------------------------------------------------
// Elementwise helpers
// ---------------------------------------------------------------------------
__global__ void fill_kernel(float* __restrict__ p, float v, long long n) {
    long long i = (long long)blockIdx.x * blockDim.x + threadIdx.x;
    if (i < n) p[i] = v;
}

__global__ void degree_kernel(const long long* __restrict__ dst,
                              float* __restrict__ deg, long long E) {
    long long i = (long long)blockIdx.x * blockDim.x + threadIdx.x;
    if (i < E) atomicAdd(&deg[(int)dst[i]], 1.0f);
}

__global__ void rsqrt_kernel(float* __restrict__ p, int n) {
    int i = blockIdx.x * blockDim.x + threadIdx.x;
    if (i < n) p[i] = rsqrtf(p[i]);
}

// agg[n,f] = relu(agg_scatter[n,f] + h[n,f]*dinv[n]^2 + bias[f]); F = power of 2
template <int F, int LOGF>
__global__ void finalize_agg(const float* __restrict__ h,
                             const float* __restrict__ dinv,
                             const float* __restrict__ bias,
                             float* __restrict__ agg, int Nn) {
    long long i = (long long)blockIdx.x * blockDim.x + threadIdx.x;
    if (i >= (long long)Nn << LOGF) return;
    const int n = (int)(i >> LOGF);
    const int f = (int)(i & (F - 1));
    const float di = dinv[n];
    const float v  = agg[i] + h[i] * di * di + bias[f];
    agg[i] = fmaxf(v, 0.0f);
}

// ---------------------------------------------------------------------------
// Edge scatter: one wave (32 lanes) per edge, lane-coalesced feature chunks.
// agg[dst] += h[src] * dinv[src] * dinv[dst]
// ---------------------------------------------------------------------------
template <int F>
__global__ __launch_bounds__(256) void scatter_edges(
    const long long* __restrict__ src, const long long* __restrict__ dst,
    const float* __restrict__ dinv, const float* __restrict__ h,
    float* __restrict__ agg, long long E) {
    const int lane = threadIdx.x & 31;
    const long long e = (long long)blockIdx.x * 8 + (threadIdx.x >> 5);
    if (e >= E) return;
    // e is wave-uniform -> force src/dst scalar so addressing goes to SALU
    int s = (int)src[e];
    int d = (int)dst[e];
    s = __builtin_amdgcn_readfirstlane(s);
    d = __builtin_amdgcn_readfirstlane(d);
    const float norm = dinv[s] * dinv[d];
    constexpr int PER = F / 32;  // floats per lane (4 for F=128, 2 for F=64)
    const float* hp = h   + (size_t)s * F + lane * PER;
    float*       ap = agg + (size_t)d * F + lane * PER;
    if constexpr (PER == 4) {
        const float4 v = *(const float4*)hp;  // one global_load_b128
        atomicAdd(ap + 0, v.x * norm);
        atomicAdd(ap + 1, v.y * norm);
        atomicAdd(ap + 2, v.z * norm);
        atomicAdd(ap + 3, v.w * norm);
    } else {
        const float2 v = *(const float2*)hp;  // one global_load_b64
        atomicAdd(ap + 0, v.x * norm);
        atomicAdd(ap + 1, v.y * norm);
    }
}

// ---------------------------------------------------------------------------
// WMMA GEMM with fragment-swizzled LDS staging.
//
// ISA 7.12.2 inverse maps (16-bit operands, wave32), tile-local k in [0,32):
//   A 16x32 (row r):  lane = r        + ((k&15)>=8 ? 16 : 0)
//   B 32x16 (col c):  lane = (c&15)   + ((k&15)>=8 ? 16 : 0)
//   both:             elem = (k>=16 ? 8 : 0) + (k&7)
// LDS holds [frag][lane][16 halves] so each lane's v16h is a contiguous,
// 32B-aligned run -> 2x ds_load_b128 per fragment.
//
// C[M,NOUT] = A[M,K] @ W[K,NOUT] (+bias / +relu per EPI: 0 none, 1 b+relu, 2 b)
// Block: 256 thr = 8 waves, 128-row tile; wave w: rows w*16..w*16+15, all NOUT.
// ---------------------------------------------------------------------------
template <int K, int NOUT, int EPI>
__global__ __launch_bounds__(256) void wmma_gemm(
    const float* __restrict__ A, const float* __restrict__ W,
    const float* __restrict__ bias, float* __restrict__ C, int M) {
    constexpr int NT = NOUT / 16;  // col tiles (also B fragments per k-step)
    constexpr int KT = K / 32;     // k steps
    __shared__ _Float16 AsF[8 * 32 * 16];    // [wave][lane][elem]  (8 KB)
    __shared__ _Float16 WsF[NT * 32 * 16];   // [tile][lane][elem]  (<= 8 KB)

    const int tid  = threadIdx.x;
    const int lane = tid & 31;
    const int wave = tid >> 5;
    const int row0 = blockIdx.x * 128;

    v8f acc[NT];
    const v8f vzero = {0.f, 0.f, 0.f, 0.f, 0.f, 0.f, 0.f, 0.f};
#pragma unroll
    for (int n = 0; n < NT; ++n) acc[n] = vzero;

    for (int kt = 0; kt < KT; ++kt) {
        const int k0 = kt * 32;

        // --- Stage A tile (128 rows x 32 k), fp32 -> f16, fragment order. ---
        // 1024 float4 groups; group (r, k4) -> 4 contiguous halves in one lane.
#pragma unroll
        for (int idx = tid; idx < 1024; idx += 256) {
            const int r  = idx >> 3;        // 0..127
            const int k4 = (idx & 7) * 4;   // 0,4,...,28 (tile-local k)
            float4 v = make_float4(0.f, 0.f, 0.f, 0.f);
            if (row0 + r < M)
                v = *(const float4*)(A + (size_t)(row0 + r) * K + k0 + k4);
            const int w    = r >> 4;
            const int ln   = (r & 15) + (((k4 & 15) >= 8) ? 16 : 0);
            const int eb   = ((k4 >= 16) ? 8 : 0) + (k4 & 7);
            v4h h4;
            h4.x = (_Float16)v.x; h4.y = (_Float16)v.y;
            h4.z = (_Float16)v.z; h4.w = (_Float16)v.w;
            *(v4h*)(AsF + ((w * 32 + ln) << 4) + eb) = h4;  // ds_store_b64
        }

        // --- Stage W tile (32 k x NOUT), fp32 -> f16, fragment order. ---
        constexpr int WCNT = 32 * NOUT / 4;
        for (int idx = tid; idx < WCNT; idx += 256) {
            const int k  = idx / (NOUT / 4);        // 0..31 (tile-local)
            const int c4 = (idx % (NOUT / 4)) * 4;  // col group (within a 16-tile)
            const float4 v = *(const float4*)(W + (size_t)(k0 + k) * NOUT + c4);
            const int n    = c4 >> 4;
            const int lnb  = (((k & 15) >= 8) ? 16 : 0);
            const int e    = ((k >= 16) ? 8 : 0) + (k & 7);
            _Float16* base = WsF + ((n * 32 + lnb + (c4 & 15)) << 4) + e;
            base[0 << 4] = (_Float16)v.x;   // ds_store_b16 (+_d16_hi) pairs
            base[1 << 4] = (_Float16)v.y;
            base[2 << 4] = (_Float16)v.z;
            base[3 << 4] = (_Float16)v.w;
        }
        __syncthreads();

        // --- Fragments are contiguous per lane: 2x ds_load_b128 each. ---
        const v16h a = *(const v16h*)(AsF + ((wave * 32 + lane) << 4));
#pragma unroll
        for (int n = 0; n < NT; ++n) {
            const v16h b = *(const v16h*)(WsF + ((n * 32 + lane) << 4));
            acc[n] = __builtin_amdgcn_wmma_f32_16x16x32_f16(
                false, a, false, b, (short)0, acc[n], false, false);
        }
        __syncthreads();
    }

    // Epilogue. C/D layout: lane col = lane&15; vgpr v is row v (+8 hi half).
    // Fast path for interior blocks (all rows valid): plain stores with
    // compile-time immediate offsets, no per-element exec-mask churn.
    const int c     = lane & 15;
    const int rbase = row0 + wave * 16 + ((lane >> 4) * 8);
    if (row0 + 128 <= M) {
#pragma unroll
        for (int n = 0; n < NT; ++n) {
            const int col = n * 16 + c;
            float* Cp = C + (size_t)rbase * NOUT + col;
#pragma unroll
            for (int v = 0; v < 8; ++v) {
                float val = acc[n][v];
                if constexpr (EPI >= 1) val += bias[col];
                if constexpr (EPI == 1) val = fmaxf(val, 0.0f);
                Cp[(size_t)v * NOUT] = val;  // global_store_b32, imm offset
            }
        }
    } else {
#pragma unroll
        for (int n = 0; n < NT; ++n) {
            const int col = n * 16 + c;
#pragma unroll
            for (int v = 0; v < 8; ++v) {
                const int r = rbase + v;
                if (r < M) {
                    float val = acc[n][v];
                    if constexpr (EPI >= 1) val += bias[col];
                    if constexpr (EPI == 1) val = fmaxf(val, 0.0f);
                    C[(size_t)r * NOUT + col] = val;
                }
            }
        }
    }
}

// ---------------------------------------------------------------------------
// Launcher
// ---------------------------------------------------------------------------
extern "C" void kernel_launch(void* const* d_in, const int* in_sizes, int n_in,
                              void* d_out, int out_size, void* d_ws, size_t ws_size,
                              hipStream_t stream) {
    const float*     x  = (const float*)d_in[0];
    const long long* ei = (const long long*)d_in[1];
    const float* W1 = (const float*)d_in[2];
    const float* b1 = (const float*)d_in[3];
    const float* W2 = (const float*)d_in[4];
    const float* b2 = (const float*)d_in[5];
    const float* W3 = (const float*)d_in[6];
    const float* b3 = (const float*)d_in[7];
    const float* W4 = (const float*)d_in[8];
    const float* b4 = (const float*)d_in[9];

    const int       N = N_NODES;
    const long long E = (long long)in_sizes[1] / 2;
    const long long* src = ei;
    const long long* dst = ei + E;

    // Workspace layout (floats): dinv[N] | bufA[N*128] | bufB[N*128]
    float* ws   = (float*)d_ws;
    float* dinv = ws;
    float* bufA = ws + N;
    float* bufB = bufA + (size_t)N * 128;
    float* h1   = bufA;                   // [N,128]
    float* agg1 = bufB;                   // [N,128]
    float* h2   = bufA;                   // [N,64]  (h1 dead)
    float* agg2 = bufA + (size_t)N * 64;  // [N,64]
    float* h3   = bufB;                   // [N,32]  (agg1 dead)
    float* out  = (float*)d_out;          // [N,16]

    const dim3 blk(256);
    const int  gemmGrid = (N + 127) / 128;

    // --- degrees + norm ---
    fill_kernel<<<(N + 255) / 256, blk, 0, stream>>>(dinv, 1.0f, N);  // self-loops
    degree_kernel<<<(unsigned)((E + 255) / 256), blk, 0, stream>>>(dst, dinv, E);
    rsqrt_kernel<<<(N + 255) / 256, blk, 0, stream>>>(dinv, N);

    // --- layer 1: h1 = x@W1 ; agg1 = relu(scatter + selfloop + b1) ---
    wmma_gemm<512, 128, 0><<<gemmGrid, blk, 0, stream>>>(x, W1, nullptr, h1, N);
    fill_kernel<<<(unsigned)(((long long)N * 128 + 255) / 256), blk, 0, stream>>>(
        agg1, 0.0f, (long long)N * 128);
    scatter_edges<128><<<(unsigned)((E + 7) / 8), blk, 0, stream>>>(src, dst, dinv, h1, agg1, E);
    finalize_agg<128, 7><<<(unsigned)(((long long)N * 128 + 255) / 256), blk, 0, stream>>>(
        h1, dinv, b1, agg1, N);

    // --- layer 2: h2 = agg1@W2 ; agg2 = relu(scatter + selfloop + b2) ---
    wmma_gemm<128, 64, 0><<<gemmGrid, blk, 0, stream>>>(agg1, W2, nullptr, h2, N);
    fill_kernel<<<(unsigned)(((long long)N * 64 + 255) / 256), blk, 0, stream>>>(
        agg2, 0.0f, (long long)N * 64);
    scatter_edges<64><<<(unsigned)((E + 7) / 8), blk, 0, stream>>>(src, dst, dinv, h2, agg2, E);
    finalize_agg<64, 6><<<(unsigned)(((long long)N * 64 + 255) / 256), blk, 0, stream>>>(
        h2, dinv, b2, agg2, N);

    // --- layer 3: h3 = relu(agg2@W3 + b3) ---
    wmma_gemm<64, 32, 1><<<gemmGrid, blk, 0, stream>>>(agg2, W3, b3, h3, N);

    // --- layer 4: out = h3@W4 + b4 ---
    wmma_gemm<32, 16, 2><<<gemmGrid, blk, 0, stream>>>(h3, W4, b4, out, N);
}
